// relation_GATs_3heads_2499670966390
// MI455X (gfx1250) — matrix operand order repaired and checked by reading, not compiled, not run on hardware
//
#include <hip/hip_runtime.h>
#include <stdint.h>

typedef _Float16 half_t;
typedef __attribute__((ext_vector_type(8)))  _Float16 v8h;
typedef __attribute__((ext_vector_type(16))) _Float16 v16h;
typedef __attribute__((ext_vector_type(8)))  float    v8f;

#define Nn 4000
#define Tt 60
#define Df 6
#define Hh 64
#define Gg 192   // 3*H

__device__ __forceinline__ int imin(int a, int b) { return a < b ? a : b; }

// ---------------- WMMA helpers (CDNA5 wave32 layouts, ISA 7.12.2) ----------
// A fragment 16x32 f16 from row-major [M][ld] source.
// Per lane the 16 halfs are two contiguous 8-half runs -> 2 x 16B vector loads.
__device__ __forceinline__ v16h load_A(const half_t* src, int ld, int lane) {
  int m  = lane & 15;
  int kb = (lane >> 4) ? 16 : 0;
  const half_t* base = src + m * ld + kb;
  v8h lo = *(const v8h*)base;          // K = kb .. kb+7
  v8h hi = *(const v8h*)(base + 8);    // K = kb+8 .. kb+15
  v16h r;
#pragma unroll
  for (int j = 0; j < 8; ++j) { r[j] = lo[j]; r[8 + j] = hi[j]; }
  return r;
}

// B fragment pre-packed per-lane contiguous: frag[lane][0..15] -> one 32B load.
__device__ __forceinline__ v16h load_Bp(const half_t* pk, int lane) {
  return *((const v16h*)pk + lane);
}

__device__ __forceinline__ v8f wmma_f16(v16h a, v16h b, v8f c) {
  return __builtin_amdgcn_wmma_f32_16x16x32_f16(false, a, false, b, (short)0, c,
                                                false, false);
}

__device__ __forceinline__ v8f vzero() {
  v8f z;
#pragma unroll
  for (int i = 0; i < 8; ++i) z[i] = 0.f;
  return z;
}
__device__ __forceinline__ v8f vadds(v8f x, float s) {
#pragma unroll
  for (int i = 0; i < 8; ++i) x[i] += s;
  return x;
}
// sigmoid / tanh via v_exp_f32 + v_rcp_f32 (no IEEE div_scale chains)
__device__ __forceinline__ v8f vsig(v8f x) {
#pragma unroll
  for (int i = 0; i < 8; ++i)
    x[i] = __builtin_amdgcn_rcpf(1.f + __expf(-x[i]));
  return x;
}
__device__ __forceinline__ v8f vtanh(v8f x) {
#pragma unroll
  for (int i = 0; i < 8; ++i)
    x[i] = 1.f - 2.f * __builtin_amdgcn_rcpf(__expf(2.f * x[i]) + 1.f);
  return x;
}
__device__ __forceinline__ v8f vmul(v8f a, v8f b) {
#pragma unroll
  for (int i = 0; i < 8; ++i) a[i] *= b[i];
  return a;
}
__device__ __forceinline__ v8f vadd(v8f a, v8f b) {
#pragma unroll
  for (int i = 0; i < 8; ++i) a[i] += b[i];
  return a;
}

// ------------- pack W [Gg][Kin] into fragment-ready f16 B tiles ------------
// frag f = tile*nch + c ; lane: n=lane&15, kb=(lane>>4)*16 ; elem j: k=c*32+kb+j
__global__ void pack_frags(const float* __restrict__ W, int Kin, int nch,
                           half_t* __restrict__ dst, int nfrag) {
  int idx = blockIdx.x * 256 + threadIdx.x;
  if (idx >= nfrag * 512) return;
  int f = idx >> 9, r = idx & 511;
  int lane = r >> 4, j = r & 15;
  int tile = f / nch, c = f % nch;
  int n = lane & 15, kb = (lane >> 4) * 16;
  int k = c * 32 + kb + j;
  float v = (k < Kin) ? W[(16 * tile + n) * Kin + k] : 0.f;
  dst[idx] = (half_t)v;
}

// ---------------- 2-layer GRU, 64 rows / block, 4 waves, WMMA core --------
__global__ __launch_bounds__(128) void gru_kernel(
    const float* __restrict__ x,
    const half_t* __restrict__ pih0, const half_t* __restrict__ phh0,
    const half_t* __restrict__ pih1, const half_t* __restrict__ phh1,
    const float* __restrict__ bih0, const float* __restrict__ bhh0,
    const float* __restrict__ bih1, const float* __restrict__ bhh1,
    float* __restrict__ hout, half_t* __restrict__ hf16out) {
  __shared__ half_t xb[64 * 32];    // x_t rows, K padded to 32
  __shared__ half_t h0s[64 * 64];   // layer0 hidden state, f16
  __shared__ half_t h1s[64 * 64];   // layer1 hidden state, f16
  __shared__ float  bb[4 * Gg];     // b_ih0, b_hh0, b_ih1, b_hh1

  const int tid  = threadIdx.x;
  const int wave = tid >> 5, lane = tid & 31;
  const int r0   = wave * 16;                // wave-local row base
  const int n0   = blockIdx.x * 64;          // global row base
  const int colb = lane & 15;
  const int rowb = (lane >> 4) * 8;

  for (int i = tid; i < 64 * 32; i += 128) xb[i] = (half_t)0.f;
  for (int i = tid; i < 64 * 64; i += 128) { h0s[i] = (half_t)0.f; h1s[i] = (half_t)0.f; }
  for (int i = tid; i < Gg; i += 128) {
    bb[i] = bih0[i]; bb[Gg + i] = bhh0[i]; bb[2 * Gg + i] = bih1[i]; bb[3 * Gg + i] = bhh1[i];
  }
  __syncthreads();

  v8f h0f[4], h1f[4];
#pragma unroll
  for (int j = 0; j < 4; ++j) { h0f[j] = vzero(); h1f[j] = vzero(); }

  for (int t = 0; t < Tt; ++t) {
    __syncthreads();
    for (int i = tid; i < 64 * Df; i += 128) {
      int r = i / Df, f = i % Df;
      int gr = imin(n0 + r, Nn - 1);
      xb[r * 32 + f] = (half_t)x[gr * (Df * Tt) + f * Tt + t];
    }
    __syncthreads();

    // ---------------- layer 0 : fused r/z/n per 16-col tile ---------------
    {
      v16h xA = load_A(xb + r0 * 32, 32, lane);
      v16h a0 = load_A(h0s + r0 * 64, 64, lane);
      v16h a1 = load_A(h0s + r0 * 64 + 32, 64, lane);
#pragma unroll
      for (int j = 0; j < 4; ++j) {
        v8f r = vzero();                               // reset gate, tile j
        r = wmma_f16(xA, load_Bp(pih0 + (size_t)j * 512, lane), r);
        r = wmma_f16(a0, load_Bp(phh0 + (size_t)(2 * j) * 512, lane), r);
        r = wmma_f16(a1, load_Bp(phh0 + (size_t)(2 * j + 1) * 512, lane), r);
        int colr = 16 * j + colb;
        r = vsig(vadds(r, bb[colr] + bb[Gg + colr]));

        v8f z = vzero();                               // update gate, tile j+4
        z = wmma_f16(xA, load_Bp(pih0 + (size_t)(j + 4) * 512, lane), z);
        z = wmma_f16(a0, load_Bp(phh0 + (size_t)(2 * (j + 4)) * 512, lane), z);
        z = wmma_f16(a1, load_Bp(phh0 + (size_t)(2 * (j + 4) + 1) * 512, lane), z);
        int colz = 16 * (j + 4) + colb;
        z = vsig(vadds(z, bb[colz] + bb[Gg + colz]));

        v8f gi = wmma_f16(xA, load_Bp(pih0 + (size_t)(j + 8) * 512, lane), vzero());
        v8f gh = wmma_f16(a0, load_Bp(phh0 + (size_t)(2 * (j + 8)) * 512, lane), vzero());
        gh = wmma_f16(a1, load_Bp(phh0 + (size_t)(2 * (j + 8) + 1) * 512, lane), gh);
        int coln = 16 * (j + 8) + colb;
        v8f nn = vtanh(vadd(vadds(gi, bb[coln]),
                            vmul(r, vadds(gh, bb[Gg + coln]))));
        v8f hn;
#pragma unroll
        for (int e = 0; e < 8; ++e) hn[e] = nn[e] + z[e] * (h0f[j][e] - nn[e]);
        h0f[j] = hn;
#pragma unroll
        for (int e = 0; e < 8; ++e)
          h0s[(r0 + rowb + e) * 64 + 16 * j + colb] = (half_t)hn[e];
      }
    }
    asm volatile("s_wait_dscnt 0x0" ::: "memory");  // wave-local LDS RAW

    // ---------------- layer 1 (input = updated h0) -------------------------
    {
      v16h i0v = load_A(h0s + r0 * 64, 64, lane);
      v16h i1v = load_A(h0s + r0 * 64 + 32, 64, lane);
      v16h p0v = load_A(h1s + r0 * 64, 64, lane);
      v16h p1v = load_A(h1s + r0 * 64 + 32, 64, lane);
#pragma unroll
      for (int j = 0; j < 4; ++j) {
        v8f r = vzero();
        r = wmma_f16(i0v, load_Bp(pih1 + (size_t)(2 * j) * 512, lane), r);
        r = wmma_f16(i1v, load_Bp(pih1 + (size_t)(2 * j + 1) * 512, lane), r);
        r = wmma_f16(p0v, load_Bp(phh1 + (size_t)(2 * j) * 512, lane), r);
        r = wmma_f16(p1v, load_Bp(phh1 + (size_t)(2 * j + 1) * 512, lane), r);
        int colr = 16 * j + colb;
        r = vsig(vadds(r, bb[2 * Gg + colr] + bb[3 * Gg + colr]));

        v8f z = vzero();
        z = wmma_f16(i0v, load_Bp(pih1 + (size_t)(2 * (j + 4)) * 512, lane), z);
        z = wmma_f16(i1v, load_Bp(pih1 + (size_t)(2 * (j + 4) + 1) * 512, lane), z);
        z = wmma_f16(p0v, load_Bp(phh1 + (size_t)(2 * (j + 4)) * 512, lane), z);
        z = wmma_f16(p1v, load_Bp(phh1 + (size_t)(2 * (j + 4) + 1) * 512, lane), z);
        int colz = 16 * (j + 4) + colb;
        z = vsig(vadds(z, bb[2 * Gg + colz] + bb[3 * Gg + colz]));

        v8f gi = wmma_f16(i0v, load_Bp(pih1 + (size_t)(2 * (j + 8)) * 512, lane), vzero());
        gi = wmma_f16(i1v, load_Bp(pih1 + (size_t)(2 * (j + 8) + 1) * 512, lane), gi);
        v8f gh = wmma_f16(p0v, load_Bp(phh1 + (size_t)(2 * (j + 8)) * 512, lane), vzero());
        gh = wmma_f16(p1v, load_Bp(phh1 + (size_t)(2 * (j + 8) + 1) * 512, lane), gh);
        int coln = 16 * (j + 8) + colb;
        v8f nn = vtanh(vadd(vadds(gi, bb[2 * Gg + coln]),
                            vmul(r, vadds(gh, bb[3 * Gg + coln]))));
        v8f hn;
#pragma unroll
        for (int e = 0; e < 8; ++e) hn[e] = nn[e] + z[e] * (h1f[j][e] - nn[e]);
        h1f[j] = hn;
#pragma unroll
        for (int e = 0; e < 8; ++e)
          h1s[(r0 + rowb + e) * 64 + 16 * j + colb] = (half_t)hn[e];
      }
    }
    asm volatile("s_wait_dscnt 0x0" ::: "memory");
  }

  // write final hidden state (layer 1)
#pragma unroll
  for (int j = 0; j < 4; ++j)
#pragma unroll
    for (int e = 0; e < 8; ++e) {
      int gr = n0 + r0 + rowb + e;
      if (gr < Nn) {
        int col = 16 * j + colb;
        hout[gr * Hh + col]    = h1f[j][e];
        hf16out[gr * Hh + col] = (half_t)h1f[j][e];
      }
    }
}

// ---------------- per-head a_i = h.W[:H], c_i = h.W[H:] --------------------
__global__ void head_prep(const float* __restrict__ h, const float* __restrict__ W0,
                          const float* __restrict__ W1, const float* __restrict__ W2,
                          float* __restrict__ a, float* __restrict__ c) {
  int i = blockIdx.x * blockDim.x + threadIdx.x;
  if (i >= Nn) return;
  float a0 = 0, c0 = 0, a1 = 0, c1 = 0, a2 = 0, c2 = 0;
  for (int d = 0; d < Hh; ++d) {
    float hv = h[i * Hh + d];
    a0 += hv * W0[d];  c0 += hv * W0[Hh + d];
    a1 += hv * W1[d];  c1 += hv * W1[Hh + d];
    a2 += hv * W2[d];  c2 += hv * W2[Hh + d];
  }
  a[i] = a0; a[Nn + i] = a1; a[2 * Nn + i] = a2;
  c[i] = c0; c[Nn + i] = c1; c[2 * Nn + i] = c2;
}

// ---------------- pass 1: per-row online (max, sum) for 3 heads ------------
__global__ __launch_bounds__(256) void stat_kernel(
    const float* __restrict__ rel, const float* __restrict__ a, const float* __restrict__ c,
    const float* __restrict__ b0, const float* __restrict__ b1, const float* __restrict__ b2,
    float* __restrict__ M, float* __restrict__ L) {
  int i    = blockIdx.x * 8 + (threadIdx.x >> 5);
  int lane = threadIdx.x & 31;
  if (i >= Nn) return;
  float bk[3] = {b0[0], b1[0], b2[0]};
  float ai[3] = {a[i], a[Nn + i], a[2 * Nn + i]};
  float mk[3] = {-3.0e38f, -3.0e38f, -3.0e38f};
  float lk[3] = {0.f, 0.f, 0.f};
  for (int j = lane; j < Nn; j += 32) {
    const float4* rp = (const float4*)(rel + ((size_t)i * Nn + j) * 8);
    float4 q0 = rp[0], q1 = rp[1];
    float mask = q0.x + q0.y + q0.z + q0.w + q1.x + q1.y + q1.z + q1.w;
#pragma unroll
    for (int k = 0; k < 3; ++k) {
      float w = ai[k] + c[k * Nn + j] + bk[k];
      w = (w > 0.f) ? w : 0.01f * w;
      float s  = (mask == 0.f) ? -1.0e6f : mask * w;
      float nm = fmaxf(mk[k], s);
      lk[k] = lk[k] * __expf(mk[k] - nm) + __expf(s - nm);
      mk[k] = nm;
    }
  }
  for (int off = 16; off > 0; off >>= 1) {
#pragma unroll
    for (int k = 0; k < 3; ++k) {
      float om = __shfl_xor(mk[k], off, 32);
      float ol = __shfl_xor(lk[k], off, 32);
      float nm = fmaxf(mk[k], om);
      lk[k] = lk[k] * __expf(mk[k] - nm) + ol * __expf(om - nm);
      mk[k] = nm;
    }
  }
  if (lane == 0) {
#pragma unroll
    for (int k = 0; k < 3; ++k) { M[i * 3 + k] = mk[k]; L[i * 3 + k] = lk[k]; }
  }
}

// ---------------- pass 2: flash-style agg = p_bar @ h via WMMA -------------
__global__ __launch_bounds__(256) void agg_kernel(
    const float* __restrict__ rel, const half_t* __restrict__ hf16,
    const float* __restrict__ a, const float* __restrict__ c,
    const float* __restrict__ b0, const float* __restrict__ b1, const float* __restrict__ b2,
    const float* __restrict__ M, const float* __restrict__ L, float* __restrict__ agg) {
  __shared__ half_t hbf[4 * 512];       // h tile in B-fragment-ready layout
  __shared__ float  cb[3 * 32];         // c_k for this j-chunk
  __shared__ half_t pbuf[8][16 * 32];   // per-wave probability tile (A matrix)
  __shared__ float  rowc[8][16][9];     // per-row: a0..2, M0..2, 1/(3*L)0..2

  const int tid  = threadIdx.x;
  const int wave = tid >> 5, lane = tid & 31;
  const int i0   = blockIdx.x * 128 + wave * 16;
  const int colb = lane & 15;
  const int rowb = (lane >> 4) * 8;
  const float bk[3] = {b0[0], b1[0], b2[0]};

  for (int idx = tid; idx < 128; idx += 256) {
    int w = idx / 16, r = idx % 16;
    int gi = imin(blockIdx.x * 128 + w * 16 + r, Nn - 1);
#pragma unroll
    for (int k = 0; k < 3; ++k) {
      rowc[w][r][k]     = a[k * Nn + gi];
      rowc[w][r][3 + k] = M[gi * 3 + k];
      rowc[w][r][6 + k] = 1.f / (3.f * L[gi * 3 + k]);
    }
  }
  __syncthreads();

  v8f acc[4];
#pragma unroll
  for (int j = 0; j < 4; ++j) acc[j] = vzero();

  const int nchunk = (Nn + 31) / 32;
  for (int ch = 0; ch < nchunk; ++ch) {
    int j0 = ch * 32;
    __syncthreads();
    // stage h tile straight into B-fragment layout: frag ct, lane = n+16*(k/16), elem = k&15
    for (int idx = tid; idx < 32 * 64; idx += 256) {
      int jr = idx >> 6, d = idx & 63;               // jr = K index, d = column
      int tile = d >> 4, n = d & 15;
      hbf[tile * 512 + (n + 16 * (jr >> 4)) * 16 + (jr & 15)] =
          hf16[(size_t)imin(j0 + jr, Nn - 1) * Hh + d];
    }
    for (int idx = tid; idx < 96; idx += 256) {
      int k = idx / 32, jr = idx % 32;
      cb[idx] = c[k * Nn + imin(j0 + jr, Nn - 1)];
    }
    __syncthreads();

    // build 16x32 probability tile for this wave's rows
#pragma unroll 1
    for (int it = 0; it < 16; ++it) {
      int gi = imin(i0 + it, Nn - 1);
      int gj = imin(j0 + lane, Nn - 1);
      const float4* rp = (const float4*)(rel + ((size_t)gi * Nn + gj) * 8);
      float4 q0 = rp[0], q1 = rp[1];
      if (j0 + 32 < Nn)
        __builtin_prefetch(rel + ((size_t)gi * Nn + (j0 + 32 + lane)) * 8, 0, 1);
      float mask = q0.x + q0.y + q0.z + q0.w + q1.x + q1.y + q1.z + q1.w;
      float pj = 0.f;
#pragma unroll
      for (int k = 0; k < 3; ++k) {
        float w = rowc[wave][it][k] + cb[k * 32 + lane] + bk[k];
        w = (w > 0.f) ? w : 0.01f * w;
        float s = (mask == 0.f) ? -1.0e6f : mask * w;
        pj += __expf(s - rowc[wave][it][3 + k]) * rowc[wave][it][6 + k];
      }
      pbuf[wave][it * 32 + lane] = (half_t)pj;
    }
    asm volatile("s_wait_dscnt 0x0" ::: "memory");  // wave-local LDS RAW

    v16h A = load_A(&pbuf[wave][0], 32, lane);
#pragma unroll
    for (int ct = 0; ct < 4; ++ct)
      acc[ct] = wmma_f16(A, load_Bp(hbf + ct * 512, lane), acc[ct]);
  }

#pragma unroll
  for (int j = 0; j < 4; ++j)
#pragma unroll
    for (int e = 0; e < 8; ++e) {
      int gi = i0 + rowb + e;
      if (gi < Nn) agg[gi * Hh + 16 * j + colb] = acc[j][e];
    }
}

// ---------------- final head: pred = [h, agg] @ fc_w.T + fc_b --------------
__global__ void pred_kernel(const float* __restrict__ h, const float* __restrict__ agg,
                            const float* __restrict__ fcw, const float* __restrict__ fcb,
                            float* __restrict__ out) {
  int i = blockIdx.x * blockDim.x + threadIdx.x;
  if (i >= Nn) return;
  float s = fcb[0];
  for (int d = 0; d < Hh; ++d)
    s += h[i * Hh + d] * fcw[d] + agg[i * Hh + d] * fcw[Hh + d];
  out[i] = s;
}

extern "C" void kernel_launch(void* const* d_in, const int* in_sizes, int n_in,
                              void* d_out, int out_size, void* d_ws, size_t ws_size,
                              hipStream_t stream) {
  const float* x    = (const float*)d_in[0];
  const float* rel  = (const float*)d_in[1];
  const float* Wih0 = (const float*)d_in[2];
  const float* Whh0 = (const float*)d_in[3];
  const float* bih0 = (const float*)d_in[4];
  const float* bhh0 = (const float*)d_in[5];
  const float* Wih1 = (const float*)d_in[6];
  const float* Whh1 = (const float*)d_in[7];
  const float* bih1 = (const float*)d_in[8];
  const float* bhh1 = (const float*)d_in[9];
  const float* W0   = (const float*)d_in[10];
  const float* b0   = (const float*)d_in[11];
  const float* W1   = (const float*)d_in[12];
  const float* b1   = (const float*)d_in[13];
  const float* W2   = (const float*)d_in[14];
  const float* b2   = (const float*)d_in[15];
  const float* fcw  = (const float*)d_in[16];
  const float* fcb  = (const float*)d_in[17];
  float* out = (float*)d_out;

  char* p = (char*)d_ws;
  half_t* pih0 = (half_t*)p;  p += (size_t)12 * 512 * sizeof(half_t);
  half_t* phh0 = (half_t*)p;  p += (size_t)24 * 512 * sizeof(half_t);
  half_t* pih1 = (half_t*)p;  p += (size_t)24 * 512 * sizeof(half_t);
  half_t* phh1 = (half_t*)p;  p += (size_t)24 * 512 * sizeof(half_t);
  p = (char*)(((uintptr_t)p + 255) & ~(uintptr_t)255);
  float*  hbuf = (float*)p;   p += (size_t)Nn * Hh * sizeof(float);
  half_t* hf   = (half_t*)p;  p += (size_t)Nn * Hh * sizeof(half_t);
  float*  abuf = (float*)p;   p += (size_t)3 * Nn * sizeof(float);
  float*  cbuf = (float*)p;   p += (size_t)3 * Nn * sizeof(float);
  float*  Mbuf = (float*)p;   p += (size_t)3 * Nn * sizeof(float);
  float*  Lbuf = (float*)p;   p += (size_t)3 * Nn * sizeof(float);
  p = (char*)(((uintptr_t)p + 255) & ~(uintptr_t)255);
  float*  aggb = (float*)p;   p += (size_t)Nn * Hh * sizeof(float);

  pack_frags<<<(12 * 512 + 255) / 256, 256, 0, stream>>>(Wih0, Df, 1, pih0, 12);
  pack_frags<<<(24 * 512 + 255) / 256, 256, 0, stream>>>(Whh0, Hh, 2, phh0, 24);
  pack_frags<<<(24 * 512 + 255) / 256, 256, 0, stream>>>(Wih1, Hh, 2, pih1, 24);
  pack_frags<<<(24 * 512 + 255) / 256, 256, 0, stream>>>(Whh1, Hh, 2, phh1, 24);
  gru_kernel<<<(Nn + 63) / 64, 128, 0, stream>>>(x, pih0, phh0, pih1, phh1,
                                                 bih0, bhh0, bih1, bhh1, hbuf, hf);
  head_prep<<<(Nn + 255) / 256, 256, 0, stream>>>(hbuf, W0, W1, W2, abuf, cbuf);
  stat_kernel<<<(Nn + 7) / 8, 256, 0, stream>>>(rel, abuf, cbuf, b0, b1, b2, Mbuf, Lbuf);
  agg_kernel<<<(Nn + 127) / 128, 256, 0, stream>>>(rel, hf, abuf, cbuf, b0, b1, b2,
                                                   Mbuf, Lbuf, aggb);
  pred_kernel<<<(Nn + 255) / 256, 256, 0, stream>>>(hbuf, aggb, fcw, fcb, out);
}
